// TransNeXt_33174327394601
// MI455X (gfx1250) — matrix operand order repaired and checked by reading, not compiled, not run on hardware
//
#include <hip/hip_runtime.h>
#include <math.h>

#define BDIM   4
#define NTOK   4096
#define CDIM   256
#define NHEADS 8
#define HDIM   32
#define PPOOL  256
#define LLOC   9
#define HRES   64
#define WRES   64
#define KDIM   256   // K of every GEMM in this net

typedef __bf16 bf16;
typedef bf16  v16bf __attribute__((ext_vector_type(16)));
typedef bf16  v4bf  __attribute__((ext_vector_type(4)));
typedef float v8f   __attribute__((ext_vector_type(8)));

// ---------------------------------------------------------------------------
// CDNA5 bf16 16x16x32 WMMA lane layouts (ISA 7.12.2):
//  A (16x32): lanes 0-15 row M=lane, K={0..7,16..23}; lanes 16-31 same rows,
//             K={8..15,24..31}.
//  B (32x16): lanes 0-15 col N=lane, K={0..7,16..23}; lanes 16-31 col N=lane-16,
//             K={8..15,24..31} (same K order as A).
//  C/D (16x16 f32): lanes 0-15 N=lane, VGPR r -> M=r; lanes 16-31 M=r+8.
// ---------------------------------------------------------------------------
__device__ __forceinline__ v16bf cvt_a_frag(float4 a0, float4 a1, float4 a2, float4 a3) {
  v16bf a;
  a[0]=(bf16)a0.x; a[1]=(bf16)a0.y; a[2]=(bf16)a0.z; a[3]=(bf16)a0.w;
  a[4]=(bf16)a1.x; a[5]=(bf16)a1.y; a[6]=(bf16)a1.z; a[7]=(bf16)a1.w;
  a[8]=(bf16)a2.x; a[9]=(bf16)a2.y; a[10]=(bf16)a2.z; a[11]=(bf16)a2.w;
  a[12]=(bf16)a3.x; a[13]=(bf16)a3.y; a[14]=(bf16)a3.z; a[15]=(bf16)a3.w;
  return a;
}

__device__ __forceinline__ float gelu_exact(float v) {
  return 0.5f * v * (1.0f + erff(v * 0.70710678118654752f));
}

// ---------------------------------------------------------------------------
// Generic GEMM: C[M x Nout] = act(A[M x K] @ W[Nout x K]^T + bias).
// Block = 256 thr (8 waves) -> 128 rows x 128 cols. W strip pre-converted to
// bf16 in LDS (64 KB) shared by all waves; A fragment double-buffered in
// float4 registers so global loads overlap the 8 WMMAs per K-step.
// Grid: (M/128, Nout/128). K must be KDIM.
// ---------------------------------------------------------------------------
__global__ void __launch_bounds__(256) wmma_gemm_bias(
    const float* __restrict__ A, const float* __restrict__ Wt,
    const float* __restrict__ bias, float* __restrict__ C,
    int M, int Nout, int K, int ldc, int act)
{
  __shared__ bf16 wsh[128 * KDIM];   // 64 KB
  const int tid   = threadIdx.x;
  const int lane  = tid & 31;
  const int wave  = tid >> 5;
  const int khalf = lane >> 4;
  const int l15   = lane & 15;
  const int rbase = blockIdx.x * 128 + wave * 16;
  const int cbase = blockIdx.y * 128;
  const int arow  = rbase + l15;

  // cooperative stage: 128 cols x K fp32 -> bf16 LDS, coalesced float4 reads
  {
    const float* wp = Wt + (size_t)cbase * K;
    for (int idx = tid * 4; idx < 128 * KDIM; idx += 256 * 4) {
      float4 f = *(const float4*)(wp + idx);
      v4bf o;
      o[0] = (bf16)f.x; o[1] = (bf16)f.y; o[2] = (bf16)f.z; o[3] = (bf16)f.w;
      *(v4bf*)&wsh[idx] = o;
    }
  }
  __syncthreads();

  v8f acc[8];
#pragma unroll
  for (int t = 0; t < 8; ++t) acc[t] = {};

  const float* arow_p = A + (size_t)arow * K;
  float4 ra0, ra1, ra2, ra3;
  {
    const float* p = arow_p + khalf * 8;
    ra0 = *(const float4*)(p);
    ra1 = *(const float4*)(p + 4);
    ra2 = *(const float4*)(p + 16);
    ra3 = *(const float4*)(p + 20);
  }

  for (int kk = 0; kk < K; kk += 32) {
    v16bf af = cvt_a_frag(ra0, ra1, ra2, ra3);
    if (kk + 32 < K) {   // prefetch next A fragment while WMMAs run
      const float* p = arow_p + kk + 32 + khalf * 8;
      ra0 = *(const float4*)(p);
      ra1 = *(const float4*)(p + 4);
      ra2 = *(const float4*)(p + 16);
      ra3 = *(const float4*)(p + 20);
    }
#pragma unroll
    for (int t = 0; t < 8; ++t) {
      const int collocal = t * 16 + l15;
      v16bf bf;
#pragma unroll
      for (int i = 0; i < 16; ++i)
        bf[i] = wsh[collocal * KDIM + kk + khalf * 8 + (i & 7) + (i >> 3) * 16];
      acc[t] = __builtin_amdgcn_wmma_f32_16x16x32_bf16(
          false, af, false, bf, (short)0, acc[t], false, false);
    }
  }

#pragma unroll
  for (int t = 0; t < 8; ++t) {
    const int col = cbase + t * 16 + l15;
    const float bv = bias[col];
#pragma unroll
    for (int r = 0; r < 8; ++r) {
      const int rr = rbase + r + khalf * 8;
      float v = acc[t][r] + bv;
      if (act) v = gelu_exact(v);
      C[(size_t)rr * ldc + col] = v;
    }
  }
}

// ---------------------------------------------------------------------------
// q/k post-processing: per-head L2 norm (wave32 = one head), q scaling,
// k normalized in place inside Y. Grid = B*N blocks, 256 thr (8 waves=8 heads)
// ---------------------------------------------------------------------------
__global__ void __launch_bounds__(256) postproc_qk(
    float* __restrict__ Y, const float* __restrict__ qe,
    const float* __restrict__ temp, const float* __restrict__ sls,
    float* __restrict__ qn, float* __restrict__ qs)
{
  const int h = threadIdx.x >> 5;
  const int d = threadIdx.x & 31;
  const int bn = blockIdx.x;
  const int b = bn / NTOK;
  const int n = bn % NTOK;
  float* rowp = Y + (size_t)bn * 768;

  float q = rowp[h * HDIM + d];
  float ss = q * q;
#pragma unroll
  for (int o = 16; o >= 1; o >>= 1) ss += __shfl_xor(ss, o);
  const float qnv = q / fmaxf(sqrtf(ss), 1e-12f);
  const float sp  = log1pf(expf(temp[h]));
  const float qsv = (qnv + qe[h * HDIM + d]) * sp * sls[n];
  const size_t qi = (((size_t)b * NHEADS + h) * NTOK + n) * HDIM + d;
  qn[qi] = qnv;
  qs[qi] = qsv;

  float k = rowp[CDIM + h * HDIM + d];
  float ks = k * k;
#pragma unroll
  for (int o = 16; o >= 1; o >>= 1) ks += __shfl_xor(ks, o);
  rowp[CDIM + h * HDIM + d] = k / fmaxf(sqrtf(ks), 1e-12f);
}

// ---------------------------------------------------------------------------
// 4x4 mean pool of gelu'd x_sr + LayerNorm over C. Grid = B*P, block 256.
// ---------------------------------------------------------------------------
__global__ void __launch_bounds__(256) pool_layernorm(
    const float* __restrict__ xsr, const float* __restrict__ g,
    const float* __restrict__ bb, float* __restrict__ xpn)
{
  __shared__ float red[256];
  const int bp = blockIdx.x;
  const int b  = bp >> 8;
  const int p  = bp & 255;
  const int ph = p >> 4, pw = p & 15;
  const int c  = threadIdx.x;

  float s = 0.f;
#pragma unroll
  for (int r = 0; r < 4; ++r)
#pragma unroll
    for (int q = 0; q < 4; ++q) {
      const int n = (ph * 4 + r) * WRES + (pw * 4 + q);
      s += xsr[((size_t)b * NTOK + n) * CDIM + c];
    }
  const float xv = s * (1.f / 16.f);

  red[c] = xv;
  __syncthreads();
  for (int o = 128; o > 0; o >>= 1) { if (c < o) red[c] += red[c + o]; __syncthreads(); }
  const float mu = red[0] * (1.f / CDIM);
  __syncthreads();
  const float dv = xv - mu;
  red[c] = dv * dv;
  __syncthreads();
  for (int o = 128; o > 0; o >>= 1) { if (c < o) red[c] += red[c + o]; __syncthreads(); }
  const float var = red[0] * (1.f / CDIM);
  xpn[(size_t)bp * CDIM + c] = dv * rsqrtf(var + 1e-5f) * g[c] + bb[c];
}

// ---------------------------------------------------------------------------
// Split kvp -> k_pool (L2-normalized per head) and v_pool, (B,H,P,HD) layout.
// ---------------------------------------------------------------------------
__global__ void __launch_bounds__(256) split_norm_pool(
    const float* __restrict__ kvp, float* __restrict__ kpool,
    float* __restrict__ vpool)
{
  const int bp = blockIdx.x;
  const int b  = bp >> 8;
  const int p  = bp & 255;
  const int h  = threadIdx.x >> 5;
  const int d  = threadIdx.x & 31;
  float k = kvp[(size_t)bp * 512 + h * HDIM + d];
  float ss = k * k;
#pragma unroll
  for (int o = 16; o >= 1; o >>= 1) ss += __shfl_xor(ss, o);
  const size_t oi = (((size_t)b * NHEADS + h) * PPOOL + p) * HDIM + d;
  kpool[oi] = k / fmaxf(sqrtf(ss), 1e-12f);
  vpool[oi] = kvp[(size_t)bp * 512 + CDIM + h * HDIM + d];
}

// ---------------------------------------------------------------------------
// CPB MLP: bias_tab[t, h] = relu(coord @ w1^T + b1) @ w2^T + b2. 1 wave / row.
// ---------------------------------------------------------------------------
__global__ void __launch_bounds__(32) cpb_mlp(
    const float* __restrict__ tab, const float* __restrict__ w1,
    const float* __restrict__ b1, const float* __restrict__ w2,
    const float* __restrict__ b2, float* __restrict__ btab, int T)
{
  const int t = blockIdx.x;
  if (t >= T) return;
  const int lane = threadIdx.x;
  const float cx = tab[(size_t)t * 2 + 0];
  const float cy = tab[(size_t)t * 2 + 1];
  float acc[NHEADS];
#pragma unroll
  for (int h = 0; h < NHEADS; ++h) acc[h] = 0.f;
  for (int j = lane; j < 512; j += 32) {
    const float h1 = fmaxf(fmaf(cx, w1[j * 2], fmaf(cy, w1[j * 2 + 1], b1[j])), 0.f);
#pragma unroll
    for (int h = 0; h < NHEADS; ++h) acc[h] += h1 * w2[h * 512 + j];
  }
#pragma unroll
  for (int h = 0; h < NHEADS; ++h) {
#pragma unroll
    for (int o = 16; o >= 1; o >>= 1) acc[h] += __shfl_xor(acc[h], o);
  }
  if (lane == 0) {
#pragma unroll
    for (int h = 0; h < NHEADS; ++h) btab[(size_t)t * NHEADS + h] = acc[h] + b2[h];
  }
}

// ---------------------------------------------------------------------------
// Fused attention: one wave per (b, h, 16-row q-tile).
//  P1: WMMA q_scaled @ k_pool^T -> 16x256 scores in LDS (async v_pool staging
//      to LDS in flight).
//  P2: local 3x3 scores + CPB bias gather + softmax over 9+256 columns.
//  P3: WMMA probs @ v_pool (1/sum folded into bf16 conversion).
//  P4: x_local + x_pool -> out_pre (B,N,C).
// ---------------------------------------------------------------------------
__global__ void __launch_bounds__(32) attn_fused(
    const float* __restrict__ qs, const float* __restrict__ qn,
    const float* __restrict__ Y,
    const float* __restrict__ kpool, const float* __restrict__ vpool,
    const float* __restrict__ btab, const int* __restrict__ ridx,
    const float* __restrict__ rbl, const float* __restrict__ tok,
    const float* __restrict__ lbias, float* __restrict__ outp)
{
  __shared__ float att[16 * PPOOL];     // 16 KB score tile
  __shared__ float vsh[PPOOL * HDIM];   // 32 KB v_pool stage
  __shared__ float rstat[16];

  const int lane  = threadIdx.x;
  const int rloc  = lane & 15;
  const int khalf = lane >> 4;
  const int tiles = NTOK / 16;
  const int bh    = blockIdx.x / tiles;
  const int tile  = blockIdx.x % tiles;
  const int b     = bh / NHEADS;
  const int h     = bh % NHEADS;
  const int row   = tile * 16 + rloc;

  // ---- async stage v_pool[b,h] (P*HD fp32 = 32 KB) into LDS ----
  {
    const unsigned lbase = (unsigned)(size_t)(void*)vsh;
    const unsigned long long gbase =
        (unsigned long long)(const void*)(vpool + (size_t)bh * PPOOL * HDIM);
#pragma unroll 4
    for (int it = 0; it < 64; ++it) {
      unsigned loff = lbase + (unsigned)(it * 512 + lane * 16);
      unsigned long long ga = gbase + (unsigned long long)(it * 512 + lane * 16);
      asm volatile("global_load_async_to_lds_b128 %0, %1, off"
                   :: "v"(loff), "v"(ga) : "memory");
    }
  }

  // ---- Phase 1: pooled-attention scores via WMMA ----
  v16bf af;
  {
    const float* qp = qs + ((size_t)bh * NTOK + row) * HDIM + khalf * 8;
    float4 a0 = *(const float4*)(qp);
    float4 a1 = *(const float4*)(qp + 4);
    float4 a2 = *(const float4*)(qp + 16);
    float4 a3 = *(const float4*)(qp + 20);
    af = cvt_a_frag(a0, a1, a2, a3);
  }
#pragma unroll
  for (int t = 0; t < 16; ++t) {
    const int p = t * 16 + rloc;
    const float* kp = kpool + ((size_t)bh * PPOOL + p) * HDIM + khalf * 8;
    float4 c0 = *(const float4*)(kp);
    float4 c1 = *(const float4*)(kp + 4);
    float4 c2 = *(const float4*)(kp + 16);
    float4 c3 = *(const float4*)(kp + 20);
    v16bf bfr = cvt_a_frag(c0, c1, c2, c3);  // same K order as A fragment
    v8f c = {};
    c = __builtin_amdgcn_wmma_f32_16x16x32_bf16(false, af, false, bfr,
                                                (short)0, c, false, false);
#pragma unroll
    for (int r = 0; r < 8; ++r)
      att[(r + khalf * 8) * PPOOL + t * 16 + rloc] = c[r];
  }
  __syncthreads();

  // ---- Phase 2: local scores, CPB bias, softmax ----
  float qsr[16], qnr[16];
  {
    const float* qp = qs + ((size_t)bh * NTOK + row) * HDIM + khalf * 16;
    const float* np = qn + ((size_t)bh * NTOK + row) * HDIM + khalf * 16;
#pragma unroll
    for (int i = 0; i < 16; ++i) { qsr[i] = qp[i]; qnr[i] = np[i]; }
  }
  const int ii = row / WRES;
  const int jj = row % WRES;
  float sl[LLOC];
  int   n2a[LLOC];
  bool  va[LLOC];
  float m = -1e30f;
#pragma unroll
  for (int l = 0; l < LLOC; ++l) {
    const int di = l / 3 - 1, dj = l % 3 - 1;
    const int i2 = ii + di, j2 = jj + dj;
    const bool v = (i2 >= 0) & (i2 < HRES) & (j2 >= 0) & (j2 < WRES);
    va[l] = v;
    const int n2 = i2 * WRES + j2;
    n2a[l] = n2;
    float part = 0.f;
    if (v) {
      const float* kp = Y + ((size_t)b * NTOK + n2) * 768 + CDIM + h * HDIM + khalf * 16;
#pragma unroll
      for (int i = 0; i < 16; ++i) part = fmaf(qsr[i], kp[i], part);
    }
    part += __shfl_xor(part, 16);
    sl[l] = v ? part + rbl[h * LLOC + l] : -1e30f;
    m = fmaxf(m, sl[l]);
  }
  const int* rp = ridx + (size_t)row * PPOOL;
  for (int p = khalf * 128; p < khalf * 128 + 128; ++p) {
    const float v = att[rloc * PPOOL + p] + btab[(size_t)rp[p] * NHEADS + h];
    att[rloc * PPOOL + p] = v;
    m = fmaxf(m, v);
  }
  m = fmaxf(m, __shfl_xor(m, 16));
  float ssum = 0.f;
  for (int p = khalf * 128; p < khalf * 128 + 128; ++p) {
    const float e = __expf(att[rloc * PPOOL + p] - m);
    att[rloc * PPOOL + p] = e;
    ssum += e;
  }
  float pl[LLOC];
#pragma unroll
  for (int l = 0; l < LLOC; ++l) pl[l] = __expf(sl[l] - m);
  if (khalf == 0) {
#pragma unroll
    for (int l = 0; l < LLOC; ++l) ssum += pl[l];
  }
  ssum += __shfl_xor(ssum, 16);
  const float rcp = 1.0f / ssum;
  if (khalf == 0) rstat[rloc] = rcp;

  // w_loc and x_local (lane pair owns one row; each half owns 16 of HD=32)
  float wl[LLOC];
#pragma unroll
  for (int l = 0; l < LLOC; ++l) {
    float td = 0.f;
#pragma unroll
    for (int i = 0; i < 16; ++i)
      td = fmaf(qnr[i], tok[((size_t)h * HDIM + khalf * 16 + i) * LLOC + l], td);
    td += __shfl_xor(td, 16);
    wl[l] = td + lbias[h * LLOC + l] + pl[l] * rcp;
  }
  float xl[16];
#pragma unroll
  for (int i = 0; i < 16; ++i) xl[i] = 0.f;
#pragma unroll
  for (int l = 0; l < LLOC; ++l) {
    if (va[l]) {
      const float* vp = Y + ((size_t)b * NTOK + n2a[l]) * 768 + 2 * CDIM + h * HDIM + khalf * 16;
#pragma unroll
      for (int i = 0; i < 16; ++i) xl[i] = fmaf(wl[l], vp[i], xl[i]);
    }
  }
  __syncthreads();
  asm volatile("s_wait_asynccnt 0" ::: "memory");
  __syncthreads();

  // ---- Phase 3: x_pool = probs @ v_pool via WMMA ----
  const float myrcp = rstat[rloc];
  v8f xa0 = {}, xa1 = {};
#pragma unroll
  for (int kk = 0; kk < PPOOL; kk += 32) {
    v16bf ap;
#pragma unroll
    for (int i = 0; i < 8; ++i)
      ap[i] = (bf16)(att[rloc * PPOOL + kk + khalf * 8 + i] * myrcp);
#pragma unroll
    for (int i = 0; i < 8; ++i)
      ap[8 + i] = (bf16)(att[rloc * PPOOL + kk + 16 + khalf * 8 + i] * myrcp);
    v16bf b0, b1;
#pragma unroll
    for (int i = 0; i < 16; ++i)
      b0[i] = (bf16)vsh[(kk + khalf * 8 + (i & 7) + (i >> 3) * 16) * HDIM + rloc];
#pragma unroll
    for (int i = 0; i < 16; ++i)
      b1[i] = (bf16)vsh[(kk + khalf * 8 + (i & 7) + (i >> 3) * 16) * HDIM + 16 + rloc];
    xa0 = __builtin_amdgcn_wmma_f32_16x16x32_bf16(false, ap, false, b0, (short)0, xa0, false, false);
    xa1 = __builtin_amdgcn_wmma_f32_16x16x32_bf16(false, ap, false, b1, (short)0, xa1, false, false);
  }
  __syncthreads();
  float* xpool = att;  // reuse the score tile
#pragma unroll
  for (int r = 0; r < 8; ++r) {
    xpool[(r + khalf * 8) * HDIM + rloc]      = xa0[r];
    xpool[(r + khalf * 8) * HDIM + 16 + rloc] = xa1[r];
  }
  __syncthreads();

  // ---- Phase 4: combine and write out_pre (B,N,C) ----
  float* op = outp + ((size_t)b * NTOK + row) * CDIM + h * HDIM + khalf * 16;
#pragma unroll
  for (int i = 0; i < 16; ++i)
    op[i] = xl[i] + xpool[rloc * HDIM + khalf * 16 + i];
}

// ---------------------------------------------------------------------------
extern "C" void kernel_launch(void* const* d_in, const int* in_sizes, int n_in,
                              void* d_out, int out_size, void* d_ws, size_t ws_size,
                              hipStream_t stream) {
  const float* x     = (const float*)d_in[0];
  const float* Wq    = (const float*)d_in[1];
  const float* bq    = (const float*)d_in[2];
  const float* qe    = (const float*)d_in[3];
  const float* temp  = (const float*)d_in[4];
  const float* Wkv   = (const float*)d_in[5];
  const float* bkv   = (const float*)d_in[6];
  const float* Wsr   = (const float*)d_in[7];
  const float* bsr   = (const float*)d_in[8];
  const float* ng    = (const float*)d_in[9];
  const float* nb    = (const float*)d_in[10];
  const float* w1    = (const float*)d_in[11];
  const float* b1    = (const float*)d_in[12];
  const float* w2    = (const float*)d_in[13];
  const float* b2    = (const float*)d_in[14];
  const float* rbl   = (const float*)d_in[15];
  const float* tok   = (const float*)d_in[16];
  const float* lbi   = (const float*)d_in[17];
  const float* Wproj = (const float*)d_in[18];
  const float* bproj = (const float*)d_in[19];
  /* d_in[20] padding_mask: recomputed analytically on-device */
  const float* sls   = (const float*)d_in[21];
  const float* rct   = (const float*)d_in[22];
  const int*   ridx  = (const int*)d_in[23];
  const int T = in_sizes[22] / 2;

  // workspace layout (fp32 elements)
  float* Y     = (float*)d_ws;                         // 16384*768  raw q|k|v
  float* qn    = Y     + (size_t)16384 * 768;          // B,H,N,HD
  float* qsb   = qn    + (size_t)BDIM * NHEADS * NTOK * HDIM;
  float* xsr   = qsb   + (size_t)BDIM * NHEADS * NTOK * HDIM; // also out_pre
  float* xpn   = xsr   + (size_t)16384 * 256;
  float* kvp   = xpn   + (size_t)1024 * 256;
  float* kpool = kvp   + (size_t)1024 * 512;
  float* vpool = kpool + (size_t)BDIM * NHEADS * PPOOL * HDIM;
  float* btab  = vpool + (size_t)BDIM * NHEADS * PPOOL * HDIM;

  cpb_mlp<<<T, 32, 0, stream>>>(rct, w1, b1, w2, b2, btab, T);
  wmma_gemm_bias<<<dim3(128, 2), 256, 0, stream>>>(x, Wq,  bq,  Y,       16384, 256, 256, 768, 0);
  wmma_gemm_bias<<<dim3(128, 4), 256, 0, stream>>>(x, Wkv, bkv, Y + 256, 16384, 512, 256, 768, 0);
  postproc_qk<<<16384, 256, 0, stream>>>(Y, qe, temp, sls, qn, qsb);
  wmma_gemm_bias<<<dim3(128, 2), 256, 0, stream>>>(x, Wsr, bsr, xsr, 16384, 256, 256, 256, 1);
  pool_layernorm<<<1024, 256, 0, stream>>>(xsr, ng, nb, xpn);
  wmma_gemm_bias<<<dim3(8, 4), 256, 0, stream>>>(xpn, Wkv, bkv, kvp, 1024, 512, 256, 512, 0);
  split_norm_pool<<<1024, 256, 0, stream>>>(kvp, kpool, vpool);
  attn_fused<<<BDIM * NHEADS * (NTOK / 16), 32, 0, stream>>>(
      qsb, qn, Y, kpool, vpool, btab, ridx, rbl, tok, lbi, xsr);
  wmma_gemm_bias<<<dim3(128, 2), 256, 0, stream>>>(xsr, Wproj, bproj, (float*)d_out,
                                                   16384, 256, 256, 256, 0);
}
